// CrossDConv_6425271075349
// MI455X (gfx1250) — matrix operand order repaired and checked by reading, not compiled, not run on hardware
//
#include <hip/hip_runtime.h>
#include <hip/hip_fp16.h>
#include <math.h>

typedef __attribute__((ext_vector_type(16))) _Float16 v16h;
typedef __attribute__((ext_vector_type(8)))  _Float16 h8;
typedef __attribute__((ext_vector_type(8)))  float    v8f;
typedef __attribute__((ext_vector_type(4)))  int      v4i;

#define NB    32
#define CIN   64
#define COUT  128
#define HW    3136      // 56*56
#define HID   16
#define K3    125       // 5^3
#define KPAD  128
#define ROWS  (COUT*CIN)  // 8192
#define BNEPS 1e-5f
#define PI_F  3.14159265358979323846f

// Async memory->LDS path (gfx1250 GLOBAL_LOAD_ASYNC_TO_LDS_*, ASYNCcnt).
#if defined(__HIP_DEVICE_COMPILE__) && __has_builtin(__builtin_amdgcn_global_load_async_to_lds_b128)
#define USE_ASYNC_LDS 1
typedef __attribute__((address_space(1))) v4i gv4i;   // global (__device__) int4
typedef __attribute__((address_space(3))) v4i lv4i;   // LDS int4
#else
#define USE_ASYNC_LDS 0
#endif

// ---------------- workspace layout (bytes) ----------------
#define OFF_H      ((size_t)0)                               // NB*HID*HW f32
#define OFF_STATS  (OFF_H + (size_t)NB*HID*HW*4)             // 32 f32 (sum,sumsq)
#define OFF_AFFINE (OFF_STATS + 128)                         // 32 f32 (scale,shift)
#define OFF_ROTVEC (OFF_AFFINE + 128)                        // NB*4 f32
#define OFF_WH     (OFF_ROTVEC + 512)                        // ROWS*KPAD f16
#define OFF_TH     (OFF_WH + (size_t)ROWS*KPAD*2)            // NB*KPAD*KPAD f16
// total ~9.6 MB

__global__ void k_zero_stats(float* stats) {
  if (threadIdx.x < 32) stats[threadIdx.x] = 0.0f;
}

// Stage 1: h[b,o,p] = sum_c x[b,c,p]*w1[o,c] + b1[o]; accumulate BN sums.
__global__ void k_hidden_stats(const float* __restrict__ x,
                               const float* __restrict__ w1,
                               const float* __restrict__ b1,
                               float* __restrict__ h,
                               float* __restrict__ stats) {
  __shared__ float s_sum[HID];
  __shared__ float s_sq[HID];
  int tid = threadIdx.x;
  if (tid < HID) { s_sum[tid] = 0.0f; s_sq[tid] = 0.0f; }
  __syncthreads();

  int gid = blockIdx.x * blockDim.x + tid;   // exactly NB*HW threads
  int b = gid / HW, p = gid % HW;
  float acc[HID];
  #pragma unroll
  for (int o = 0; o < HID; ++o) acc[o] = b1[o];
  const float* xp = x + (size_t)b * CIN * HW + p;
  for (int c = 0; c < CIN; ++c) {
    float xv = xp[(size_t)c * HW];
    #pragma unroll
    for (int o = 0; o < HID; ++o) acc[o] = fmaf(xv, w1[o * CIN + c], acc[o]);
  }
  float* hp = h + (size_t)b * HID * HW + p;
  #pragma unroll
  for (int o = 0; o < HID; ++o) {
    hp[(size_t)o * HW] = acc[o];
    atomicAdd(&s_sum[o], acc[o]);
    atomicAdd(&s_sq[o], acc[o] * acc[o]);
  }
  __syncthreads();
  if (tid < HID) {
    atomicAdd(&stats[tid], s_sum[tid]);
    atomicAdd(&stats[HID + tid], s_sq[tid]);
  }
}

// Stage 2: BN affine: scale = gamma*rsqrt(var+eps), shift = beta - mean*scale
__global__ void k_finalize(const float* __restrict__ stats,
                           const float* __restrict__ gamma,
                           const float* __restrict__ beta,
                           float* __restrict__ affine) {
  int c = threadIdx.x;
  if (c < HID) {
    const float n = (float)((size_t)NB * HW);
    float mean = stats[c] / n;
    float var  = stats[HID + c] / n - mean * mean;
    float inv  = rsqrtf(var + BNEPS);
    float sc   = gamma[c] * inv;
    affine[c] = sc;
    affine[HID + c] = beta[c] - mean * sc;
  }
}

// Stage 3: per (b,j): rot_vec = sum(rm * softmax(rm)) over 3136 positions
__global__ __launch_bounds__(256)
void k_rotvec(const float* __restrict__ h, const float* __restrict__ affine,
              const float* __restrict__ w2, const float* __restrict__ b2,
              float* __restrict__ rotvec) {
  __shared__ float rm[HW];
  __shared__ float red[256];
  int b = blockIdx.x, j = blockIdx.y;
  int tid = threadIdx.x;
  float sc[HID], sh[HID], wj[HID];
  #pragma unroll
  for (int c = 0; c < HID; ++c) {
    sc[c] = affine[c]; sh[c] = affine[HID + c]; wj[c] = w2[j * HID + c];
  }
  float bj = b2[j];
  float lmax = -1e30f;
  for (int p = tid; p < HW; p += 256) {
    const float* hp = h + (size_t)b * HID * HW + p;
    float v = bj;
    #pragma unroll
    for (int c = 0; c < HID; ++c) {
      float hv = fmaf(hp[(size_t)c * HW], sc[c], sh[c]);
      hv = fmaxf(hv, 0.0f);
      v = fmaf(hv, wj[c], v);
    }
    rm[p] = v;
    lmax = fmaxf(lmax, v);
  }
  red[tid] = lmax; __syncthreads();
  for (int s = 128; s > 0; s >>= 1) {
    if (tid < s) red[tid] = fmaxf(red[tid], red[tid + s]);
    __syncthreads();
  }
  float mx = red[0]; __syncthreads();
  float se = 0.0f, swe = 0.0f;
  for (int p = tid; p < HW; p += 256) {
    float v = rm[p];
    float e = __expf(v - mx);
    se += e; swe += v * e;
  }
  red[tid] = se; __syncthreads();
  for (int s = 128; s > 0; s >>= 1) { if (tid < s) red[tid] += red[tid + s]; __syncthreads(); }
  se = red[0]; __syncthreads();
  red[tid] = swe; __syncthreads();
  for (int s = 128; s > 0; s >>= 1) { if (tid < s) red[tid] += red[tid + s]; __syncthreads(); }
  swe = red[0];
  if (tid == 0) rotvec[b * 4 + j] = swe / se;
}

// Stage 4: build per-batch 125x125 separable Dirichlet matrix T_b, stored
// row-major [n = xyz][k = uvw], zero-padded to 128x128, f16.
__global__ void k_build_T(const float* __restrict__ rotvec, _Float16* __restrict__ Th) {
  int b = blockIdx.x;
  float kx = rotvec[b * 4 + 0], ky = rotvec[b * 4 + 1], kz = rotvec[b * 4 + 2];
  float ang = rotvec[b * 4 + 3];
  float nrm = sqrtf(kx * kx + ky * ky + kz * kz) + 1e-8f;
  kx /= nrm; ky /= nrm; kz /= nrm;
  ang = tanhf(ang) * (PI_F * 0.25f);
  // a = column sums of R = I + ang*K
  float a0 = 1.0f + ang * (kz - ky);
  float a1 = 1.0f + ang * (kx - kz);
  float a2 = 1.0f + ang * (ky - kx);
  float px[5], py[5], pz[5];
  #pragma unroll
  for (int m = 0; m < 5; ++m) {
    float t0 = (2.0f * PI_F / 5.0f) * ((float)m - a0);
    float t1 = (2.0f * PI_F / 5.0f) * ((float)m - a1);
    float t2 = (2.0f * PI_F / 5.0f) * ((float)m - a2);
    px[m] = (1.0f + 2.0f * __cosf(t0) + 2.0f * __cosf(2.0f * t0)) * 0.2f;
    py[m] = (1.0f + 2.0f * __cosf(t1) + 2.0f * __cosf(2.0f * t1)) * 0.2f;
    pz[m] = (1.0f + 2.0f * __cosf(t2) + 2.0f * __cosf(2.0f * t2)) * 0.2f;
  }
  _Float16* T = Th + (size_t)b * KPAD * KPAD;
  for (int idx = threadIdx.x; idx < KPAD * KPAD; idx += 256) {
    int n = idx >> 7, k = idx & 127;
    float v = 0.0f;
    if (n < K3 && k < K3) {
      int x = n / 25, y = (n / 5) % 5, z = n % 5;
      int u = k / 25, vv = (k / 5) % 5, w = k % 5;
      v = px[(x - u + 5) % 5] * py[(y - vv + 5) % 5] * pz[(z - w + 5) % 5];
    }
    T[idx] = (_Float16)v;
  }
}

// Stage 5: convert weights (8192 x 125 f32) -> f16 padded to K=128
__global__ void k_convW(const float* __restrict__ w3, _Float16* __restrict__ Wh) {
  int idx = blockIdx.x * 256 + threadIdx.x;   // ROWS*KPAD total
  int r = idx >> 7, k = idx & 127;
  Wh[idx] = (_Float16)(k < K3 ? w3[(size_t)r * K3 + k] : 0.0f);
}

// Stage 6: per-batch GEMM  Out_b(8192x125) = W(8192x128) * T_b^T  via WMMA.
__global__ __launch_bounds__(256, 1)
void k_wmma_gemm(const _Float16* __restrict__ Wh, const _Float16* __restrict__ Th,
                 float* __restrict__ out) {
  __shared__ __align__(16) _Float16 Asm[128 * KPAD];   // 32 KB: 128 rows of W
  __shared__ __align__(16) _Float16 Bsm[KPAD * KPAD];  // 32 KB: T_b row-major [n][k]
  int b  = blockIdx.y;
  int m0 = blockIdx.x * 128;
  int tid = threadIdx.x;

  // ---- cooperative tile fills (2048 x 16B per tile / 256 threads) ----
  const uint4* srcA = (const uint4*)(Wh + (size_t)m0 * KPAD);
  const uint4* srcB = (const uint4*)(Th + (size_t)b * KPAD * KPAD);
  uint4* dA = (uint4*)Asm;
  uint4* dB = (uint4*)Bsm;
#if USE_ASYNC_LDS
  {
    #pragma unroll
    for (int i = 0; i < 8; ++i) {
      // builtin signature (from clang diagnostic): (int4 __device__*, int4 LDS*, imm, imm)
      __builtin_amdgcn_global_load_async_to_lds_b128(
          (gv4i*)(v4i*)(uint4*)&srcA[tid + 256 * i],
          (lv4i*)(v4i*)&dA[tid + 256 * i], 0, 0);
      __builtin_amdgcn_global_load_async_to_lds_b128(
          (gv4i*)(v4i*)(uint4*)&srcB[tid + 256 * i],
          (lv4i*)(v4i*)&dB[tid + 256 * i], 0, 0);
    }
#if __has_builtin(__builtin_amdgcn_s_wait_asynccnt)
    __builtin_amdgcn_s_wait_asynccnt(0);
#else
    asm volatile("s_wait_asynccnt 0" ::: "memory");
#endif
  }
#else
  #pragma unroll
  for (int i = 0; i < 8; ++i) {
    dA[tid + 256 * i] = srcA[tid + 256 * i];
    dB[tid + 256 * i] = srcB[tid + 256 * i];
  }
#endif
  __syncthreads();

  int wave = tid >> 5, lane = tid & 31;
  int mrow = lane & 15;        // A: M index ; B/C: N index
  int hi   = lane >> 4;        // which K/M half this lane covers
  const _Float16* arow = &Asm[(wave * 16 + mrow) * KPAD];

  union V16 { v16h v; h8 h[2]; };
  union V8F { v8f v; float e[8]; };

  for (int nt = 0; nt < 8; ++nt) {
    v8f c = {};
    #pragma unroll
    for (int kk = 0; kk < 4; ++kk) {
      V16 av, bv;
      // A 16x32 f16 layout (ISA 7.12.2): lane<16 -> K 0..7 & 16..23; lane>=16 -> K 8..15 & 24..31
      const _Float16* ap = arow + kk * 32 + hi * 8;
      av.h[0] = *(const h8*)(ap);
      av.h[1] = *(const h8*)(ap + 16);
      // B 32x16 f16: lane<16 holds K 0..7 & 16..23 of column N, lane>=16 holds K 8..15 & 24..31
      const _Float16* bp = &Bsm[(nt * 16 + mrow) * KPAD + kk * 32 + hi * 8];
      bv.h[0] = *(const h8*)(bp);
      bv.h[1] = *(const h8*)(bp + 16);
      c = __builtin_amdgcn_wmma_f32_16x16x32_f16(
              /*neg_a=*/false, av.v, /*neg_b=*/false, bv.v,
              /*c_mod=*/(short)0, c, /*reuse_a=*/false, /*reuse_b=*/false);
    }
    int col = nt * 16 + mrow;
    if (col < K3) {
      V8F r; r.v = c;
      // C layout: VGPR rr -> M = rr (lanes 0-15) or 8+rr (lanes 16-31).
      // Output is a 131MB streaming write; use non-temporal stores so it does
      // not evict the heavily reused Wh/Th operand set from L2.
      float* op = out + ((size_t)b * ROWS + m0 + wave * 16 + hi * 8) * (size_t)K3 + col;
      #pragma unroll
      for (int rr = 0; rr < 8; ++rr)
        __builtin_nontemporal_store(r.e[rr], op + (size_t)rr * K3);
    }
  }
}

extern "C" void kernel_launch(void* const* d_in, const int* in_sizes, int n_in,
                              void* d_out, int out_size, void* d_ws, size_t ws_size,
                              hipStream_t stream) {
  const float* x     = (const float*)d_in[0];
  const float* w3    = (const float*)d_in[1];
  const float* w1    = (const float*)d_in[2];
  const float* b1    = (const float*)d_in[3];
  const float* gamma = (const float*)d_in[4];
  const float* beta  = (const float*)d_in[5];
  const float* w2    = (const float*)d_in[6];
  const float* b2    = (const float*)d_in[7];
  float* out = (float*)d_out;

  char* ws = (char*)d_ws;
  float*    h      = (float*)(ws + OFF_H);
  float*    stats  = (float*)(ws + OFF_STATS);
  float*    affine = (float*)(ws + OFF_AFFINE);
  float*    rotvec = (float*)(ws + OFF_ROTVEC);
  _Float16* Wh     = (_Float16*)(ws + OFF_WH);
  _Float16* Th     = (_Float16*)(ws + OFF_TH);

  k_zero_stats<<<1, 64, 0, stream>>>(stats);
  k_hidden_stats<<<(NB * HW) / 256, 256, 0, stream>>>(x, w1, b1, h, stats);
  k_finalize<<<1, 32, 0, stream>>>(stats, gamma, beta, affine);
  k_rotvec<<<dim3(NB, 4), 256, 0, stream>>>(h, affine, w2, b2, rotvec);
  k_build_T<<<NB, 256, 0, stream>>>(rotvec, Th);
  k_convW<<<(ROWS * KPAD) / 256, 256, 0, stream>>>(w3, Wh);
  k_wmma_gemm<<<dim3(ROWS / 128, NB), 256, 0, stream>>>(Wh, Th, out);
}
